// OptimizedMambaGNN_64381559767401
// MI455X (gfx1250) — matrix operand order repaired and checked by reading, not compiled, not run on hardware
//
#include <hip/hip_runtime.h>
#include <hip/hip_bf16.h>
#include <math.h>

// ---------------- CDNA5 WMMA types ----------------
typedef __attribute__((ext_vector_type(16))) _Float16 v16h;
typedef __attribute__((ext_vector_type(8)))  float    v8f;
typedef __attribute__((ext_vector_type(4)))  float    v4f;
typedef __attribute__((ext_vector_type(4)))  _Float16 v4h;

#define BATCHN 2048
#define NPATCH 14
#define DMODEL 192
#define DINNER 384
#define DSTATE 16
#define DTRANK 12
#define NHEADS 8
#define DHEAD  24
#define MROWS  (BATCHN * NPATCH)   // 28672

// =====================================================================
// Weight prep: BT[Npad][Kpad] (f16) = transpose(W[K][N] fp32), zero-padded
// so the GEMM fast path never needs bounds checks on B.
// =====================================================================
__global__ void convert_wT_kernel(const float* __restrict__ W,
                                  _Float16* __restrict__ BT,
                                  int K, int N, int Kp, int Np)
{
  const int idx = blockIdx.x * blockDim.x + threadIdx.x;
  if (idx >= Kp * Np) return;
  const int n = idx % Np, k = idx / Np;
  const float v = (k < K && n < N) ? W[(size_t)k * N + n] : 0.f;
  BT[(size_t)n * Kp + k] = (_Float16)v;
}

// =====================================================================
// Fused GEMM: C[M,N] = epi( A[M,*](lda,f32) * BT[Npad,Kpad](f16) + bias + add )
// One wave per 16x16 tile, 4 waves/block, v_wmma_f32_16x16x32_f16.
// Preconditions (guaranteed by caller): M % 64 == 0, lda >= Kp, BT padded.
// Inner loop: 4x b128 A loads + pk-cvt + 2x b128 B loads + WMMA, no EXEC ops.
// epi: 0 none, 1 sigmoid, 2 softplus
// =====================================================================
__global__ __launch_bounds__(128) void gemm_wmma_f16(
    const float* __restrict__ A, int lda,
    const _Float16* __restrict__ BT,
    float* __restrict__ C, int ldc,
    const float* __restrict__ bias,
    const float* __restrict__ addsrc, int ldadd,
    int M, int N, int Kp, int epi)
{
  const int lane = threadIdx.x & 31;
  const int wave = threadIdx.x >> 5;
  const int n0 = blockIdx.x * 16;
  const int m0 = (blockIdx.y * 4 + wave) * 16;

  const int half = lane >> 4;
  const int lm   = lane & 15;
  const int arow = m0 + lm;            // A row this lane feeds (< M always)
  const int bcol = n0 + lm;            // B/C column this lane feeds (< Npad)

  v8f acc = {0.f, 0.f, 0.f, 0.f, 0.f, 0.f, 0.f, 0.f};

  const float*    ap0 = A  + (size_t)arow * lda + half * 16;
  const _Float16* bp0 = BT + (size_t)bcol * Kp  + half * 16;

#pragma unroll 2
  for (int k0 = 0; k0 < Kp; k0 += 32) {
    // ---- A fragment: K run [half*16, half*16+16) (16 floats) ----
    const float* ap = ap0 + k0;
    const v4f a0 = *(const v4f*)(ap);
    const v4f a1 = *(const v4f*)(ap + 4);
    const v4f a2 = *(const v4f*)(ap + 8);
    const v4f a3 = *(const v4f*)(ap + 12);
    // ---- B fragment: K run [half*16, half*16+16) of padded column ----
    const _Float16* bp = bp0 + k0;
    const v4h b0 = *(const v4h*)(bp);
    const v4h b1 = *(const v4h*)(bp + 4);
    const v4h b2 = *(const v4h*)(bp + 8);
    const v4h b3 = *(const v4h*)(bp + 12);

    v16h af, bf;
#pragma unroll
    for (int e = 0; e < 4; ++e) {
      af[e]      = (_Float16)a0[e];
      af[4 + e]  = (_Float16)a1[e];
      af[8 + e]  = (_Float16)a2[e];
      af[12 + e] = (_Float16)a3[e];
      bf[e]      = b0[e];
      bf[4 + e]  = b1[e];
      bf[8 + e]  = b2[e];
      bf[12 + e] = b3[e];
    }
    acc = __builtin_amdgcn_wmma_f32_16x16x32_f16(
        false, af, false, bf, (short)0, acc, false, false);
  }

  if (bcol >= N) return;               // divergence only after all WMMAs
#pragma unroll
  for (int r = 0; r < 8; ++r) {
    const int row = m0 + half * 8 + r;
    float v = acc[r];
    if (bias)   v += bias[bcol];
    if (addsrc) v += addsrc[(size_t)row * ldadd + bcol];
    if (epi == 1)      v = 1.f / (1.f + expf(-v));
    else if (epi == 2) v = (v > 20.f) ? v : log1pf(expf(v));
    C[(size_t)row * ldc + bcol] = v;
  }
}

// =====================================================================
// Patch embedding (pre-LN): out[row,c] = 0.1*x . patch_w + patch_b + pos
// =====================================================================
__global__ void patch_embed_kernel(const float* __restrict__ x,
                                   const float* __restrict__ pw,
                                   const float* __restrict__ pb,
                                   const float* __restrict__ pos,
                                   float* __restrict__ out)
{
  const int row = blockIdx.x;          // b*14 + p
  const int b = row / NPATCH, p = row % NPATCH;
  const int c = threadIdx.x;
  const float* xs = x + (size_t)b * 700 + p * 50;
  float acc = pb[c];
#pragma unroll 5
  for (int t = 0; t < 50; ++t) acc += xs[t] * 0.1f * pw[t * DMODEL + c];
  acc += pos[p * DMODEL + c];
  out[(size_t)row * DMODEL + c] = acc;
}

// =====================================================================
// LayerNorm over last dim D. EPI: 0 plain, 1 gelu, 2 gelu*scale[row/14]
// =====================================================================
template <int D, int EPI>
__global__ void ln_kernel(const float* __restrict__ x,
                          const float* __restrict__ g,
                          const float* __restrict__ b,
                          const float* __restrict__ scale,
                          float* __restrict__ y, int rows)
{
  constexpr int P = (D <= 256) ? 256 : (D <= 512 ? 512 : 1024);
  __shared__ float buf[P];
  const int row = blockIdx.x;
  const int t = threadIdx.x;
  const float v = x[(size_t)row * D + t];

  buf[t] = v; __syncthreads();
  for (int s = P / 2; s > 0; s >>= 1) {
    if (t < s && t + s < D) buf[t] += buf[t + s];
    __syncthreads();
  }
  const float mean = buf[0] * (1.f / D); __syncthreads();
  const float d = v - mean;
  buf[t] = d * d; __syncthreads();
  for (int s = P / 2; s > 0; s >>= 1) {
    if (t < s && t + s < D) buf[t] += buf[t + s];
    __syncthreads();
  }
  const float var = buf[0] * (1.f / D);
  float o = d * rsqrtf(var + 1e-5f) * g[t] + b[t];
  if constexpr (EPI >= 1) o = 0.5f * o * (1.f + erff(o * 0.70710678118f));
  if constexpr (EPI == 2) o *= scale[(size_t)(row / NPATCH) * D + t];
  y[(size_t)row * D + t] = o;
}

// =====================================================================
// Depthwise causal conv (4 taps, left pad 3) + SiLU.
// =====================================================================
__global__ void conv_silu_kernel(const float* __restrict__ uz,
                                 const float* __restrict__ cw,
                                 const float* __restrict__ cb,
                                 float* __restrict__ u)
{
  const int idx = blockIdx.x * blockDim.x + threadIdx.x;
  if (idx >= BATCHN * DINNER) return;
  const int c = idx % DINNER, b = idx / DINNER;
  const float w0 = cw[c], w1 = cw[DINNER + c], w2 = cw[2 * DINNER + c],
              w3 = cw[3 * DINNER + c], bias = cb[c];
  for (int t = 0; t < NPATCH; ++t) {
    float s = bias;
    if (t >= 3) s += uz[(size_t)(b * NPATCH + t - 3) * 768 + c] * w0;
    if (t >= 2) s += uz[(size_t)(b * NPATCH + t - 2) * 768 + c] * w1;
    if (t >= 1) s += uz[(size_t)(b * NPATCH + t - 1) * 768 + c] * w2;
    s += uz[(size_t)(b * NPATCH + t) * 768 + c] * w3;
    u[(size_t)(b * NPATCH + t) * DINNER + c] = s / (1.f + expf(-s));
  }
}

// =====================================================================
// Selective scan + skip(D) + SiLU(z) gate.
// =====================================================================
__global__ void scan_kernel(const float* __restrict__ delta,
                            const float* __restrict__ pr,
                            const float* __restrict__ ucv,
                            const float* __restrict__ uz,
                            const float* __restrict__ Alog,
                            const float* __restrict__ Dp,
                            float* __restrict__ y)
{
  const int idx = blockIdx.x * blockDim.x + threadIdx.x;
  if (idx >= BATCHN * DINNER) return;
  const int c = idx % DINNER, b = idx / DINNER;
  float A[DSTATE], st[DSTATE];
#pragma unroll
  for (int s = 0; s < DSTATE; ++s) { A[s] = -expf(Alog[c * DSTATE + s]); st[s] = 0.f; }
  const float Dv = Dp[c];
  for (int t = 0; t < NPATCH; ++t) {
    const size_t r = (size_t)(b * NPATCH + t);
    const float d  = delta[r * DINNER + c];
    const float uu = ucv[r * DINNER + c];
    const float du = d * uu;
    float yv = 0.f;
#pragma unroll
    for (int s = 0; s < DSTATE; ++s) {
      const float Bt = pr[r * 44 + DTRANK + s];
      const float Ct = pr[r * 44 + DTRANK + DSTATE + s];
      st[s] = expf(d * A[s]) * st[s] + du * Bt;
      yv += st[s] * Ct;
    }
    yv += uu * Dv;
    const float z = uz[r * 768 + DINNER + c];
    y[r * DINNER + c] = yv * (z / (1.f + expf(-z)));
  }
}

// =====================================================================
// kNN adjacency: cosine sim 14x14, keep top-9 values per row.
// =====================================================================
__global__ void knn_adj_kernel(const float* __restrict__ h, float* __restrict__ adj)
{
  __shared__ float sim[NPATCH][NPATCH];
  __shared__ float nrm[NPATCH];
  const int b = blockIdx.x, t = threadIdx.x;
  if (t < NPATCH) {
    const float* r = h + (size_t)(b * NPATCH + t) * DMODEL;
    float s = 0.f;
    for (int c = 0; c < DMODEL; ++c) s += r[c] * r[c];
    nrm[t] = fmaxf(sqrtf(s), 1e-12f);
  }
  __syncthreads();
  if (t < NPATCH * NPATCH) {
    const int n = t / NPATCH, m = t % NPATCH;
    const float* rn = h + (size_t)(b * NPATCH + n) * DMODEL;
    const float* rm = h + (size_t)(b * NPATCH + m) * DMODEL;
    float s = 0.f;
    for (int c = 0; c < DMODEL; ++c) s += rn[c] * rm[c];
    sim[n][m] = s / (nrm[n] * nrm[m]);
  }
  __syncthreads();
  if (t < NPATCH) {
    float row[NPATCH]; bool used[NPATCH];
    for (int j = 0; j < NPATCH; ++j) { row[j] = 0.f; used[j] = false; }
    for (int k = 0; k < 9; ++k) {      // K_NEI + 1
      int bi = 0; float bv = -1e30f;
      for (int j = 0; j < NPATCH; ++j)
        if (!used[j] && sim[t][j] > bv) { bv = sim[t][j]; bi = j; }
      used[bi] = true; row[bi] = bv;
    }
    for (int j = 0; j < NPATCH; ++j)
      adj[(size_t)(b * NPATCH + t) * NPATCH + j] = row[j];
  }
}

// =====================================================================
// GAT attention: block per batch, thread per (node, head). hg += elu(...)
// =====================================================================
__global__ void gat_attn_kernel(const float* __restrict__ Wh,
                                const float* __restrict__ adj,
                                const float* __restrict__ asrc,
                                const float* __restrict__ adst,
                                const float* __restrict__ bias,
                                float* __restrict__ hg)
{
  __shared__ float W[NPATCH * DMODEL];
  __shared__ float S[NPATCH][NHEADS];
  __shared__ float T[NPATCH][NHEADS];
  const int b = blockIdx.x, t = threadIdx.x;
  for (int i = t; i < NPATCH * DMODEL; i += blockDim.x)
    W[i] = Wh[(size_t)b * NPATCH * DMODEL + i];
  __syncthreads();
  if (t < NPATCH * NHEADS) {
    const int i = t / NHEADS, hh = t % NHEADS;
    float s = 0.f, tt = 0.f;
    for (int d = 0; d < DHEAD; ++d) {
      const float w = W[i * DMODEL + hh * DHEAD + d];
      s += w * asrc[hh * DHEAD + d];
      tt += w * adst[hh * DHEAD + d];
    }
    S[i][hh] = s; T[i][hh] = tt;
  }
  __syncthreads();
  if (t < NPATCH * NHEADS) {
    const int i = t / NHEADS, hh = t % NHEADS;
    float e[NPATCH], mx = -1e30f;
    for (int j = 0; j < NPATCH; ++j) {
      float v = S[i][hh] + T[j][hh];
      v = (v > 0.f) ? v : 0.2f * v;                 // leaky_relu 0.2
      v = (adj[(size_t)(b * NPATCH + i) * NPATCH + j] > 0.f) ? v : -1e9f;
      e[j] = v; mx = fmaxf(mx, v);
    }
    float sum = 0.f;
    for (int j = 0; j < NPATCH; ++j) { e[j] = expf(e[j] - mx); sum += e[j]; }
    const float inv = 1.f / sum;
    float acc[DHEAD];
    for (int d = 0; d < DHEAD; ++d) acc[d] = 0.f;
    for (int j = 0; j < NPATCH; ++j) {
      const float a = e[j] * inv;
      for (int d = 0; d < DHEAD; ++d) acc[d] += a * W[j * DMODEL + hh * DHEAD + d];
    }
    for (int d = 0; d < DHEAD; ++d) {
      const int cc = hh * DHEAD + d;
      float o = acc[d] + bias[cc];
      o = (o > 0.f) ? o : (expf(o) - 1.f);          // elu
      hg[(size_t)(b * NPATCH + i) * DMODEL + cc] += o;
    }
  }
}

// comb = [h, hg]; cm = comb.mean(tokens)
__global__ void concat_mean_kernel(const float* __restrict__ h,
                                   const float* __restrict__ hg,
                                   float* __restrict__ comb,
                                   float* __restrict__ cm)
{
  const int idx = blockIdx.x * blockDim.x + threadIdx.x;
  if (idx >= BATCHN * 2 * DMODEL) return;
  const int c = idx % (2 * DMODEL), b = idx / (2 * DMODEL);
  const float* src = (c < DMODEL) ? h : hg;
  const int cc = c % DMODEL;
  float s = 0.f;
  for (int t = 0; t < NPATCH; ++t) {
    const float v = src[(size_t)(b * NPATCH + t) * DMODEL + cc];
    comb[(size_t)(b * NPATCH + t) * 2 * DMODEL + c] = v;
    s += v;
  }
  cm[(size_t)b * 2 * DMODEL + c] = s * (1.f / NPATCH);
}

__global__ void mean_tokens_kernel(const float* __restrict__ x, float* __restrict__ out, int D)
{
  const int idx = blockIdx.x * blockDim.x + threadIdx.x;
  if (idx >= BATCHN * D) return;
  const int c = idx % D, b = idx / D;
  float s = 0.f;
  for (int t = 0; t < NPATCH; ++t) s += x[(size_t)(b * NPATCH + t) * D + c];
  out[(size_t)b * D + c] = s * (1.f / NPATCH);
}

// attention pooling: hglob[b] = softmax(q.k/sqrt(D)) @ hf
__global__ void attn_pool_kernel(const float* __restrict__ q,
                                 const float* __restrict__ hf,
                                 const float* __restrict__ k,
                                 float* __restrict__ hglob)
{
  __shared__ float sc[NPATCH];
  const int b = blockIdx.x, t = threadIdx.x;
  if (t < NPATCH) {
    float s = 0.f;
    for (int c = 0; c < DMODEL; ++c)
      s += q[(size_t)b * DMODEL + c] * k[(size_t)(b * NPATCH + t) * DMODEL + c];
    sc[t] = s * 0.0721687836f;   // 1/sqrt(192)
  }
  __syncthreads();
  float mx = -1e30f;
  for (int j = 0; j < NPATCH; ++j) mx = fmaxf(mx, sc[j]);
  float w[NPATCH], sum = 0.f;
  for (int j = 0; j < NPATCH; ++j) { w[j] = expf(sc[j] - mx); sum += w[j]; }
  const float inv = 1.f / sum;
  float acc = 0.f;
  for (int j = 0; j < NPATCH; ++j)
    acc += w[j] * hf[(size_t)(b * NPATCH + j) * DMODEL + t];
  hglob[(size_t)b * DMODEL + t] = acc * inv;
}

// =====================================================================
// host-side orchestration
// =====================================================================
static inline void gemm(hipStream_t s, const float* A, int lda, const float* W,
                        _Float16* wt, float* C, int ldc,
                        const float* bias, const float* add, int ldadd,
                        int M, int N, int K, int epi)
{
  const int Kp = (K + 31) & ~31;       // lda >= Kp holds for every call site
  const int Np = (N + 15) & ~15;
  const int tot = Kp * Np;
  convert_wT_kernel<<<(tot + 255) / 256, 256, 0, s>>>(W, wt, K, N, Kp, Np);
  dim3 g(Np / 16, M / 64);             // M % 64 == 0 for every call site
  gemm_wmma_f16<<<g, 128, 0, s>>>(A, lda, wt, C, ldc, bias, add, ldadd, M, N, Kp, epi);
}

extern "C" void kernel_launch(void* const* d_in, const int* in_sizes, int n_in,
                              void* d_out, int out_size, void* d_ws, size_t ws_size,
                              hipStream_t stream)
{
  (void)in_sizes; (void)n_in; (void)out_size; (void)ws_size;
  int ii = 0;
  auto nxt = [&]() { return (const float*)d_in[ii++]; };

  const float* x       = nxt();
  const float* patch_w = nxt();
  const float* patch_b = nxt();
  const float* pos     = nxt();
  const float* in_g    = nxt();
  const float* in_b    = nxt();
  struct MB { const float *ng,*nb,*in_w,*in_b,*conv_w,*conv_b,*xproj_w,*dt_w,*dt_b,*A_log,*D,*out_w,*out_b; } mb[4];
  for (int l = 0; l < 4; ++l) {
    mb[l].ng = nxt(); mb[l].nb = nxt(); mb[l].in_w = nxt(); mb[l].in_b = nxt();
    mb[l].conv_w = nxt(); mb[l].conv_b = nxt(); mb[l].xproj_w = nxt();
    mb[l].dt_w = nxt(); mb[l].dt_b = nxt(); mb[l].A_log = nxt(); mb[l].D = nxt();
    mb[l].out_w = nxt(); mb[l].out_b = nxt();
  }
  struct GP { const float *W,*a_src,*a_dst,*b; } gp[3];
  for (int l = 0; l < 3; ++l) { gp[l].W = nxt(); gp[l].a_src = nxt(); gp[l].a_dst = nxt(); gp[l].b = nxt(); }
  const float* ch_w = nxt(); const float* ch_b = nxt();
  const float* fus_w = nxt(); const float* fus_b = nxt();
  const float* fus_g = nxt(); const float* fus_bn = nxt();
  const float* q_w = nxt(); const float* q_b = nxt();
  const float* k_w = nxt(); const float* k_b = nxt();
  const float* c1_w = nxt(); const float* c1_b = nxt();
  const float* c1_g = nxt(); const float* c1_bn = nxt();
  const float* c2_w = nxt(); const float* c2_b = nxt();
  const float* c2_g = nxt(); const float* c2_bn = nxt();
  const float* c3_w = nxt(); const float* c3_b = nxt();

  // ---- workspace layout (floats) ----
  float* ws = (float*)d_ws;
  size_t off = 0;
  auto alloc = [&](size_t n) { float* p = ws + off; off += n; return p; };
  const size_t M = MROWS;
  float* h   = alloc(M * DMODEL);
  float* hg  = alloc(M * DMODEL);
  float* tA  = alloc(M * DMODEL);   // xln / Wh / pre-LN fusion temp
  float* R1  = alloc(M * 768);      // uz during mamba; comb+hf+k afterwards
  float* ucv = alloc(M * DINNER);
  float* pr  = alloc(M * 44);
  float* dl  = alloc(M * DINNER);
  float* yb  = alloc(M * DINNER);
  float* adj = alloc((size_t)BATCHN * NPATCH * NPATCH);
  float* cm  = alloc((size_t)BATCHN * 2 * DMODEL);
  float* cwv = alloc((size_t)BATCHN * DMODEL);
  float* hfm = alloc((size_t)BATCHN * DMODEL);
  float* qb  = alloc((size_t)BATCHN * DMODEL);
  float* hgl = alloc((size_t)BATCHN * DMODEL);
  float* z1p = alloc((size_t)BATCHN * 2 * DMODEL);
  float* z1  = alloc((size_t)BATCHN * 2 * DMODEL);
  float* z2p = alloc((size_t)BATCHN * 768);
  float* z2  = alloc((size_t)BATCHN * 768);
  float* wtf = alloc((size_t)768 * 768 / 2 + 64); // padded f16 BT scratch
  _Float16* wt = (_Float16*)wtf;
  float* comb = R1;                 // M*384 (post-mamba reuse)
  float* hf   = R1 + M * 384;       // M*192
  float* kbuf = R1 + M * 576;       // M*192

  const int EW = 256;
  const int nBC = (BATCHN * DINNER + EW - 1) / EW;
  const int nCM = (BATCHN * 2 * DMODEL + EW - 1) / EW;
  const int nMT = (BATCHN * DMODEL + EW - 1) / EW;

  // ---- 1. patch embed + input LN ----
  patch_embed_kernel<<<(int)M, DMODEL, 0, stream>>>(x, patch_w, patch_b, pos, tA);
  ln_kernel<DMODEL, 0><<<(int)M, DMODEL, 0, stream>>>(tA, in_g, in_b, nullptr, h, (int)M);

  // ---- 2. four Mamba blocks ----
  for (int l = 0; l < 4; ++l) {
    ln_kernel<DMODEL, 0><<<(int)M, DMODEL, 0, stream>>>(h, mb[l].ng, mb[l].nb, nullptr, tA, (int)M);
    gemm(stream, tA, DMODEL, mb[l].in_w, wt, R1, 768, mb[l].in_b, nullptr, 0, (int)M, 768, DMODEL, 0);
    conv_silu_kernel<<<nBC, EW, 0, stream>>>(R1, mb[l].conv_w, mb[l].conv_b, ucv);
    gemm(stream, ucv, DINNER, mb[l].xproj_w, wt, pr, 44, nullptr, nullptr, 0, (int)M, 44, DINNER, 0);
    // K=12 -> Kp=32: A reads pr cols [12,32) (finite B/C values) x zero-padded BT
    gemm(stream, pr, 44, mb[l].dt_w, wt, dl, DINNER, mb[l].dt_b, nullptr, 0, (int)M, DINNER, DTRANK, 2);
    scan_kernel<<<nBC, EW, 0, stream>>>(dl, pr, ucv, R1, mb[l].A_log, mb[l].D, yb);
    gemm(stream, yb, DINNER, mb[l].out_w, wt, h, DMODEL, mb[l].out_b, h, DMODEL, (int)M, DMODEL, DINNER, 0);
  }

  // ---- 3. kNN adjacency ----
  knn_adj_kernel<<<BATCHN, 224, 0, stream>>>(h, adj);
  hipMemcpyAsync(hg, h, M * DMODEL * sizeof(float), hipMemcpyDeviceToDevice, stream);

  // ---- 4. three GAT layers ----
  for (int l = 0; l < 3; ++l) {
    gemm(stream, hg, DMODEL, gp[l].W, wt, tA, DMODEL, nullptr, nullptr, 0, (int)M, DMODEL, DMODEL, 0);
    gat_attn_kernel<<<BATCHN, 128, 0, stream>>>(tA, adj, gp[l].a_src, gp[l].a_dst, gp[l].b, hg);
  }

  // ---- 5. fusion ----
  concat_mean_kernel<<<nCM, EW, 0, stream>>>(h, hg, comb, cm);
  gemm(stream, cm, 2 * DMODEL, ch_w, wt, cwv, DMODEL, ch_b, nullptr, 0, BATCHN, DMODEL, 2 * DMODEL, 1);
  gemm(stream, comb, 2 * DMODEL, fus_w, wt, tA, DMODEL, fus_b, nullptr, 0, (int)M, DMODEL, 2 * DMODEL, 0);
  ln_kernel<DMODEL, 2><<<(int)M, DMODEL, 0, stream>>>(tA, fus_g, fus_bn, cwv, hf, (int)M);

  // ---- 6. attention pooling ----
  mean_tokens_kernel<<<nMT, EW, 0, stream>>>(hf, hfm, DMODEL);
  gemm(stream, hfm, DMODEL, q_w, wt, qb, DMODEL, q_b, nullptr, 0, BATCHN, DMODEL, DMODEL, 0);
  gemm(stream, hf, DMODEL, k_w, wt, kbuf, DMODEL, k_b, nullptr, 0, (int)M, DMODEL, DMODEL, 0);
  attn_pool_kernel<<<BATCHN, DMODEL, 0, stream>>>(qb, hf, kbuf, hgl);

  // ---- 7. classifier head ----
  gemm(stream, hgl, DMODEL, c1_w, wt, z1p, 2 * DMODEL, c1_b, nullptr, 0, BATCHN, 2 * DMODEL, DMODEL, 0);
  ln_kernel<2 * DMODEL, 1><<<BATCHN, 2 * DMODEL, 0, stream>>>(z1p, c1_g, c1_bn, nullptr, z1, BATCHN);
  gemm(stream, z1, 2 * DMODEL, c2_w, wt, z2p, 768, c2_b, nullptr, 0, BATCHN, 768, 2 * DMODEL, 0);
  ln_kernel<768, 1><<<BATCHN, 768, 0, stream>>>(z2p, c2_g, c2_bn, nullptr, z2, BATCHN);
  gemm(stream, z2, 768, c3_w, wt, (float*)d_out, 256, c3_b, nullptr, 0, BATCHN, 256, 768, 0);
}
